// attentionlayer_5634997092873
// MI455X (gfx1250) — compile-verified, matrix operand order checked by reading, not run on hardware
//
#include <hip/hip_runtime.h>
#include <hip/hip_bf16.h>

typedef __attribute__((ext_vector_type(16))) _Float16 v16h;
typedef __attribute__((ext_vector_type(8)))  _Float16 v8h;
typedef __attribute__((ext_vector_type(8)))  float    v8f;
typedef __attribute__((ext_vector_type(4)))  float    v4f;

#define B_  16
#define C_  512
#define H_  32
#define W_  32
#define N_  1024
#define CQ_ 32
#define HP_ 34

// Build a 16x32 (A) or 32x16 (B) f16 WMMA fragment. `p` = per-lane pointer to
// the K-contiguous row (A: M-row) or column (B: N-column). CDNA5 layout:
// lanes 0-15 hold K {0..7,16..23}, lanes 16-31 hold K {8..15,24..31};
// caller passes koff = (lane>>4)*8.
__device__ __forceinline__ v16h load_frag16(const _Float16* p, int koff) {
  v8h lo = *(const v8h*)(p + koff);
  v8h hi = *(const v8h*)(p + koff + 16);
  return __builtin_shufflevector(lo, hi, 0,1,2,3,4,5,6,7,8,9,10,11,12,13,14,15);
}

// CDNA5 async copy: 16B global -> LDS per lane, tracked by ASYNCcnt. The LDS
// destination VGPR takes the wave-relative LDS byte address (low 32 bits of
// the generic shared-memory pointer in the LDS aperture).
__device__ __forceinline__ void async_copy16(_Float16* lds_dst,
                                             const _Float16* gsrc) {
  unsigned lds = (unsigned)(size_t)lds_dst;
  unsigned long long ga = (unsigned long long)(size_t)gsrc;
  asm volatile("global_load_async_to_lds_b128 %0, %1, off"
               :: "v"(lds), "v"(ga) : "memory");
}

__device__ __forceinline__ void wait_async0() {
  asm volatile("s_wait_asynccnt 0x0" ::: "memory");
}

// ---------------------------------------------------------------------------
// Pack kernels: f32 NCHW -> zero-padded channels-last f16 (B,34,34,C), and
// OIHW f32 weights -> (oc, tap, ic) f16.
// ---------------------------------------------------------------------------
__global__ void pack_pad_kernel(const float* __restrict__ src,
                                _Float16* __restrict__ dst) {
  size_t idx = (size_t)blockIdx.x * 256 + threadIdx.x;
  if (idx >= (size_t)B_ * HP_ * HP_ * C_) return;
  int c  = (int)(idx & (C_ - 1));
  size_t r = idx >> 9;
  int wp = (int)(r % HP_); r /= HP_;
  int hp = (int)(r % HP_);
  int b  = (int)(r / HP_);
  float val = 0.0f;
  if (hp >= 1 && hp <= H_ && wp >= 1 && wp <= W_)
    val = src[(((size_t)(b * C_ + c)) * H_ + (hp - 1)) * W_ + (wp - 1)];
  dst[idx] = (_Float16)val;
}

__global__ void pack_w_kernel(const float* __restrict__ src,
                              _Float16* __restrict__ dst, int total) {
  int idx = blockIdx.x * 256 + threadIdx.x;
  if (idx >= total) return;
  int ic  = idx & (C_ - 1);
  int r   = idx >> 9;
  int tap = r % 9;
  int oc  = r / 9;
  dst[idx] = (_Float16)src[((size_t)(oc * C_ + ic)) * 9 + tap];
}

// ---------------------------------------------------------------------------
// conv q/k: implicit GEMM, M=32 (oc), N = one (b,h) row of 32 w, K = 512*9.
// 8 waves: [which(q/k)][oc half][w half]. LDS double-buffered via async
// global->LDS copies so the next ic-chunk streams in under the WMMAs.
// Output layout: (b, n, 32) f16  (transposed, for contiguous attention frags).
// ---------------------------------------------------------------------------
__global__ __launch_bounds__(256) void conv_qk_kernel(
    const _Float16* __restrict__ xpad, const _Float16* __restrict__ ypad,
    const _Float16* __restrict__ wq_t, const _Float16* __restrict__ wk_t,
    const float* __restrict__ bq, const float* __restrict__ bk,
    _Float16* __restrict__ qt, _Float16* __restrict__ kt) {
  __shared__ __align__(16) _Float16 xs[2][2][3][34][40]; // [dblbuf][x/y][...]

  const int tid  = threadIdx.x;
  const int lane = tid & 31;
  const int wid  = tid >> 5;
  const int b = blockIdx.x >> 5, h = blockIdx.x & 31;

  const int which = wid >> 2;            // 0 = q (x,wq), 1 = k (y,wk)
  const int oc0   = ((wid >> 1) & 1) * 16;
  const int w0l   = (wid & 1) * 16;

  const int rowA = lane & 15;            // A: M row / B: N column
  const int koff = (lane >> 4) * 8;
  const int ncol = lane & 15;
  const int mb   = (lane >> 4) * 8;

  const _Float16* wt   = which ? wk_t : wq_t;
  const float*    bias = which ? bk : bq;
  _Float16*       outt = which ? kt : qt;

  auto stage = [&](int buf, int ic0) {
    for (int t = tid; t < 816; t += 256) {      // both x and y tiles
      int tb = t >= 408;
      int tt = t - tb * 408;
      int part = tt & 3, slot = tt >> 2;
      int dh = slot / 34, w = slot % 34;
      const _Float16* sp = (tb ? ypad : xpad) +
          (((size_t)(b * HP_ + h + dh) * HP_ + w) * C_) + ic0 + part * 8;
      async_copy16(&xs[buf][tb][dh][w][part * 8], sp);
    }
  };

  v8f acc = {};
  stage(0, 0);
  wait_async0();
  __syncthreads();

  for (int ic0 = 0; ic0 < C_; ic0 += 32) {
    const int buf = (ic0 >> 5) & 1;
    if (ic0 + 32 < C_) stage(buf ^ 1, ic0 + 32);  // overlap with compute
#pragma unroll
    for (int kh = 0; kh < 3; ++kh)
#pragma unroll
      for (int kw = 0; kw < 3; ++kw) {
        const _Float16* abase =
            wt + ((size_t)(oc0 + rowA) * 9 + kh * 3 + kw) * C_ + ic0;
        v16h a = load_frag16(abase, koff);
        v16h bm = load_frag16(&xs[buf][which][kh][w0l + kw + ncol][0], koff);
        acc = __builtin_amdgcn_wmma_f32_16x16x32_f16(
            false, a, false, bm, (short)0, acc, false, false);
      }
    wait_async0();
    __syncthreads();
  }
  // bias + transposed store: outt[(b,n,oc)] as one 16B chunk per lane
  _Float16* op = outt + ((size_t)(b * N_ + h * W_ + w0l + ncol)) * CQ_ + oc0 + mb;
  v8h o;
#pragma unroll
  for (int g = 0; g < 8; ++g)
    o[g] = (_Float16)(acc[g] + bias[oc0 + mb + g]);
  *(v8h*)op = o;
}

// ---------------------------------------------------------------------------
// conv v: implicit GEMM, M=512 (oc, gridDim.y*64 tiling), N = one (b,h) row.
// 8 waves: 4 oc-waves x 2 w-waves. Async double-buffered LDS staging.
// Output layout: (b, c, n) f16 (K-contiguous for attention GEMM2's B).
// ---------------------------------------------------------------------------
__global__ __launch_bounds__(256) void conv_v_kernel(
    const _Float16* __restrict__ ypad, const _Float16* __restrict__ wv_t,
    const float* __restrict__ bv, _Float16* __restrict__ vout) {
  __shared__ __align__(16) _Float16 xs[2][3][34][40];

  const int tid  = threadIdx.x;
  const int lane = tid & 31;
  const int wid  = tid >> 5;
  const int b = blockIdx.x >> 5, h = blockIdx.x & 31;
  const int ocTile = blockIdx.y * 64 + (wid & 3) * 16;
  const int w0l    = (wid >> 2) * 16;

  const int rowA = lane & 15;
  const int koff = (lane >> 4) * 8;
  const int ncol = lane & 15;
  const int mb   = (lane >> 4) * 8;

  auto stage = [&](int buf, int ic0) {
    for (int t = tid; t < 408; t += 256) {
      int part = t & 3, slot = t >> 2;
      int dh = slot / 34, w = slot % 34;
      const _Float16* sp = ypad +
          (((size_t)(b * HP_ + h + dh) * HP_ + w) * C_) + ic0 + part * 8;
      async_copy16(&xs[buf][dh][w][part * 8], sp);
    }
  };

  v8f acc = {};
  if (tid == 0)  // warm the weight stream for this oc tile into L2/WGP$
    __builtin_prefetch(wv_t + (size_t)ocTile * 9 * C_, 0, 0);
  stage(0, 0);
  wait_async0();
  __syncthreads();

  for (int ic0 = 0; ic0 < C_; ic0 += 32) {
    const int buf = (ic0 >> 5) & 1;
    if (ic0 + 32 < C_) stage(buf ^ 1, ic0 + 32);  // overlap with compute
#pragma unroll
    for (int kh = 0; kh < 3; ++kh)
#pragma unroll
      for (int kw = 0; kw < 3; ++kw) {
        const _Float16* abase =
            wv_t + ((size_t)(ocTile + rowA) * 9 + kh * 3 + kw) * C_ + ic0;
        v16h a = load_frag16(abase, koff);
        v16h bm = load_frag16(&xs[buf][kh][w0l + kw + ncol][0], koff);
        acc = __builtin_amdgcn_wmma_f32_16x16x32_f16(
            false, a, false, bm, (short)0, acc, false, false);
      }
    wait_async0();
    __syncthreads();
  }
  _Float16* op = vout + ((size_t)(b * C_ + ocTile + mb)) * N_ + h * W_ + w0l + ncol;
#pragma unroll
  for (int g = 0; g < 8; ++g)
    op[(size_t)g * N_] = (_Float16)(acc[g] + bv[ocTile + mb + g]);
}

// ---------------------------------------------------------------------------
// attention: one block per (b, 16-row i-block). 8 waves.
//  GEMM1: S[16][1024] = qT(16x32) @ k(32x1024), one WMMA per 16x16 tile.
//  softmax over j in LDS (f32).
//  GEMM2: outT(16x512) = P(16x1024) @ vT -> d_out (b, c, i) f32.
// ---------------------------------------------------------------------------
__global__ __launch_bounds__(256) void attn_kernel(
    const _Float16* __restrict__ qt, const _Float16* __restrict__ kt,
    const _Float16* __restrict__ vv, float* __restrict__ out) {
  __shared__ __align__(16) float S[16][1028];   // stride 1028 -> 16B-aligned rows
  __shared__ float red[16][16];

  const int tid  = threadIdx.x;
  const int lane = tid & 31;
  const int wid  = tid >> 5;
  const int b  = blockIdx.x >> 6;
  const int i0 = (blockIdx.x & 63) * 16;

  const int rowA = lane & 15;
  const int koff = (lane >> 4) * 8;
  const int ncol = lane & 15;
  const int mb   = (lane >> 4) * 8;

  // ---- GEMM1: energy ----
  const _Float16* qbase = qt + ((size_t)(b * N_ + i0 + rowA)) * CQ_;
  v16h aq = load_frag16(qbase, koff);
#pragma unroll
  for (int t = 0; t < 8; ++t) {
    int jt = wid * 8 + t;
    const _Float16* kbase = kt + ((size_t)(b * N_ + jt * 16 + rowA)) * CQ_;
    v16h bk = load_frag16(kbase, koff);
    v8f s = {};
    s = __builtin_amdgcn_wmma_f32_16x16x32_f16(
        false, aq, false, bk, (short)0, s, false, false);
#pragma unroll
    for (int g = 0; g < 8; ++g)
      S[mb + g][jt * 16 + ncol] = s[g];
  }
  __syncthreads();

  // ---- softmax: 16 threads per row, 64 columns each ----
  const int row = tid >> 4, sub = tid & 15;
  float m = -1e30f;
  for (int c = sub; c < N_; c += 16) m = fmaxf(m, S[row][c]);
  red[row][sub] = m;
  __syncthreads();
  float rm = red[row][0];
#pragma unroll
  for (int t = 1; t < 16; ++t) rm = fmaxf(rm, red[row][t]);
  __syncthreads();
  float sum = 0.0f;
  for (int c = sub; c < N_; c += 16) {
    float e = __expf(S[row][c] - rm);
    S[row][c] = e;
    sum += e;
  }
  red[row][sub] = sum;
  __syncthreads();
  float rs = 0.0f;
#pragma unroll
  for (int t = 0; t < 16; ++t) rs += red[row][t];
  const float inv = 1.0f / rs;
  for (int c = sub; c < N_; c += 16) S[row][c] *= inv;
  __syncthreads();

  // ---- GEMM2: out ----
#pragma unroll
  for (int t = 0; t < 4; ++t) {
    int ct = wid * 4 + t;
    v8f acc = {};
    for (int j0 = 0; j0 < N_; j0 += 32) {
      const float* sp = &S[rowA][j0 + koff];
      v16h a;
#pragma unroll
      for (int hh = 0; hh < 8; ++hh) a[hh] = (_Float16)sp[hh];
#pragma unroll
      for (int hh = 0; hh < 8; ++hh) a[8 + hh] = (_Float16)sp[16 + hh];
      const _Float16* vb = vv + ((size_t)(b * C_ + ct * 16 + ncol)) * N_ + j0;
      v16h bm = load_frag16(vb, koff);
      acc = __builtin_amdgcn_wmma_f32_16x16x32_f16(
          false, a, false, bm, (short)0, acc, false, false);
    }
    float* op = out + ((size_t)(b * C_ + ct * 16 + ncol)) * N_ + i0 + mb;
    *(v4f*)op       = (v4f){acc[0], acc[1], acc[2], acc[3]};
    *(v4f*)(op + 4) = (v4f){acc[4], acc[5], acc[6], acc[7]};
  }
}

// ---------------------------------------------------------------------------
extern "C" void kernel_launch(void* const* d_in, const int* in_sizes, int n_in,
                              void* d_out, int out_size, void* d_ws, size_t ws_size,
                              hipStream_t stream) {
  (void)in_sizes; (void)n_in; (void)out_size; (void)ws_size;
  const float* x  = (const float*)d_in[0];
  const float* y  = (const float*)d_in[1];
  const float* wq = (const float*)d_in[2];
  const float* bq = (const float*)d_in[3];
  const float* wk = (const float*)d_in[4];
  const float* bk = (const float*)d_in[5];
  const float* wv = (const float*)d_in[6];
  const float* bv = (const float*)d_in[7];
  float* out = (float*)d_out;

  char* ws = (char*)d_ws;
  size_t o = 0;
  auto alloc = [&](size_t bytes) {
    char* p = ws + o;
    o = (o + bytes + 255) & ~(size_t)255;
    return p;
  };
  const size_t padElems = (size_t)B_ * HP_ * HP_ * C_;
  _Float16* xpad = (_Float16*)alloc(padElems * 2);
  _Float16* ypad = (_Float16*)alloc(padElems * 2);
  _Float16* wq_t = (_Float16*)alloc((size_t)CQ_ * 9 * C_ * 2);
  _Float16* wk_t = (_Float16*)alloc((size_t)CQ_ * 9 * C_ * 2);
  _Float16* wv_t = (_Float16*)alloc((size_t)C_ * 9 * C_ * 2);
  _Float16* qtb  = (_Float16*)alloc((size_t)B_ * N_ * CQ_ * 2);
  _Float16* ktb  = (_Float16*)alloc((size_t)B_ * N_ * CQ_ * 2);
  _Float16* v16  = (_Float16*)alloc((size_t)B_ * C_ * N_ * 2);

  const int padBlocks = (int)((padElems + 255) / 256);
  pack_pad_kernel<<<padBlocks, 256, 0, stream>>>(x, xpad);
  pack_pad_kernel<<<padBlocks, 256, 0, stream>>>(y, ypad);
  pack_w_kernel<<<(CQ_ * 9 * C_ + 255) / 256, 256, 0, stream>>>(wq, wq_t, CQ_ * 9 * C_);
  pack_w_kernel<<<(CQ_ * 9 * C_ + 255) / 256, 256, 0, stream>>>(wk, wk_t, CQ_ * 9 * C_);
  pack_w_kernel<<<(C_ * 9 * C_ + 255) / 256, 256, 0, stream>>>(wv, wv_t, C_ * 9 * C_);

  conv_qk_kernel<<<B_ * H_, 256, 0, stream>>>(xpad, ypad, wq_t, wk_t, bq, bk, qtb, ktb);
  conv_v_kernel<<<dim3(B_ * H_, C_ / 64), 256, 0, stream>>>(ypad, wv_t, bv, v16);
  attn_kernel<<<B_ * 64, 256, 0, stream>>>(qtb, ktb, v16, out);
}
